// Multihead_self_attention_9895604650622
// MI455X (gfx1250) — matrix operand order repaired and checked by reading, not compiled, not run on hardware
//
#include <hip/hip_runtime.h>
#include <hip/hip_fp16.h>

// ---------------------------------------------------------------------------
// MI455X (gfx1250) fused causal MHSA with RoPE, f16 WMMA / f32 accumulate.
//   x[2,2048,1024] f32, W* [1024,1024] f32  ->  out[2,2048,1024] f32
//   1) convert x, Wq, Wk, Wv, Wo to f16
//   2) qkv_rope: block stages 32x1024 A-panel via global_load_async_to_lds_b128;
//      per-wave 32x64 strip, ping-pong double-buffered v_wmma_f32_16x16x32_f16
//      (chunk-0 WMMAs use inline C=0 -> no accumulator init movs);
//      RoPE fused in-register; V stored [B,H,dk,T].
//   3) flash attention: stage-rotated pipeline -- V(kc)/K(kc+1) loads issued
//      over softmax VALU of chunk kc; S(kc+1) WMMAs at loop bottom.
//   4) out-proj: LDS-staged ping-pong GEMM, f32 output.
// ---------------------------------------------------------------------------

typedef _Float16 h16 __attribute__((ext_vector_type(16)));
typedef _Float16 h8  __attribute__((ext_vector_type(8)));
typedef float    f8  __attribute__((ext_vector_type(8)));

#define T_SEQ   2048
#define D_MODEL 1024
#define NHEAD   16
#define DK      64
#define BT      4096

#define APAD    8
#define AROW    (D_MODEL + APAD)   // LDS A-panel row stride (halves)

__device__ __forceinline__ h16 comb16(h8 lo, h8 hi2) {
  return __builtin_shufflevector(lo, hi2, 0,1,2,3,4,5,6,7,8,9,10,11,12,13,14,15);
}

__device__ __forceinline__ f8 wmma_f16(h16 a, h16 b, f8 c) {
  return __builtin_amdgcn_wmma_f32_16x16x32_f16(false, a, false, b,
                                                (short)0, c, false, false);
}

__device__ __forceinline__ f8 f8zero() {
  f8 z = {0.f,0.f,0.f,0.f,0.f,0.f,0.f,0.f};
  return z;
}

__device__ __forceinline__ unsigned lds_off_u32(const void* p) {
  return (unsigned)(unsigned long long)p;   // low 32b of generic addr = LDS off
}

__device__ __forceinline__ void async_g2l_b128(unsigned lds_byte_off,
                                               const void* gaddr) {
  asm volatile("global_load_async_to_lds_b128 %0, %1, off"
               :: "v"(lds_byte_off), "v"(gaddr) : "memory");
}
__device__ __forceinline__ void async_wait_all() {
  asm volatile("s_wait_asynccnt 0" ::: "memory");
}

// --------------------------- 1) f32 -> f16 convert -------------------------
__global__ void __launch_bounds__(256)
cvt_f32_f16(const float* __restrict__ src, _Float16* __restrict__ dst, int n) {
  int i = (blockIdx.x * 256 + threadIdx.x) * 8;
  if (i + 7 < n) {
#pragma unroll
    for (int j = 0; j < 8; ++j) dst[i + j] = (_Float16)src[i + j];
  }
}

// ---- cooperative async stage of a 32x1024 f16 panel into LDS ----
__device__ __forceinline__ void stage_panel(_Float16* panel,
                                            const _Float16* gsrc,
                                            int tid) {
  const int row = tid >> 2;
  const int seg = tid & 3;
  const _Float16* g = gsrc + (size_t)row * D_MODEL + seg * 256;
  unsigned l = lds_off_u32(&panel[row * AROW + seg * 256]);
#pragma unroll
  for (int it = 0; it < 32; ++it)
    async_g2l_b128(l + it * 16, g + it * 8);
  async_wait_all();
  __syncthreads();
}

// ---- GEMM fragment helpers (ping-pong, no rotation copies) ----
__device__ __forceinline__ void load_ab(h16& A0, h16& A1, h16 (&B)[4],
                                        const _Float16* lp0,
                                        const _Float16* lp1,
                                        const _Float16* wb, int kb) {
  A0 = comb16(*(const h8*)(lp0 + kb), *(const h8*)(lp0 + kb + 8));
  A1 = comb16(*(const h8*)(lp1 + kb), *(const h8*)(lp1 + kb + 8));
#pragma unroll
  for (int nt = 0; nt < 4; ++nt)
    B[nt] = *(const h16*)(wb + (size_t)nt * 16 * D_MODEL + kb);
}

__device__ __forceinline__ void mma8(f8 (&acc)[2][4], h16 A0, h16 A1,
                                     const h16 (&B)[4]) {
#pragma unroll
  for (int nt = 0; nt < 4; ++nt) {
    acc[0][nt] = wmma_f16(A0, B[nt], acc[0][nt]);
    acc[1][nt] = wmma_f16(A1, B[nt], acc[1][nt]);
  }
}

// first K-chunk: C = inline 0 (initializes accumulators, no mov burst)
__device__ __forceinline__ void mma8_init(f8 (&acc)[2][4], h16 A0, h16 A1,
                                          const h16 (&B)[4]) {
#pragma unroll
  for (int nt = 0; nt < 4; ++nt) {
    acc[0][nt] = wmma_f16(A0, B[nt], f8zero());
    acc[1][nt] = wmma_f16(A1, B[nt], f8zero());
  }
}

// K-loop over 32 chunks (kb = c*32): ping-pong X/Y buffer sets.
// Schedule: load X(0); load Y(1), mmaI X(0);
//           for c=2..30 step 2 { load X(c), mma Y(c-1); load Y(c+1), mma X(c) }
//           mma Y(31).
__device__ __forceinline__ void gemm_32xk(f8 (&acc)[2][4],
                                          const _Float16* lp0,
                                          const _Float16* lp1,
                                          const _Float16* wb) {
  h16 xa0, xa1, ya0, ya1;
  h16 xb[4], yb[4];
  load_ab(xa0, xa1, xb, lp0, lp1, wb, 0);
  load_ab(ya0, ya1, yb, lp0, lp1, wb, 32);
  mma8_init(acc, xa0, xa1, xb);
  for (int c = 2; c <= 30; c += 2) {
    load_ab(xa0, xa1, xb, lp0, lp1, wb, c * 32);
    mma8(acc, ya0, ya1, yb);
    load_ab(ya0, ya1, yb, lp0, lp1, wb, (c + 1) * 32);
    mma8(acc, xa0, xa1, xb);
  }
  mma8(acc, ya0, ya1, yb);
}

// --------------------- 2) QKV projection + fused RoPE ----------------------
// grid (128, 12), block 128 (4 waves).
__global__ void __launch_bounds__(128)
qkv_rope_kernel(const _Float16* __restrict__ xh,
                const _Float16* __restrict__ wq,
                const _Float16* __restrict__ wk,
                const _Float16* __restrict__ wv,
                _Float16* __restrict__ qh,   // [B,H,T,dk]
                _Float16* __restrict__ kh,   // [B,H,T,dk]
                _Float16* __restrict__ vth)  // [B,H,dk,T]
{
  __shared__ __align__(16) _Float16 apanel[32 * AROW];

  const int tid   = threadIdx.x;
  const int lane  = tid & 31;
  const int wid   = tid >> 5;
  const int mt    = blockIdx.x;
  const int combo = blockIdx.y * 4 + wid;
  const int which = combo >> 4;
  const int head  = combo & 15;

  const int hh    = lane >> 4;
  const int lmod  = lane & 15;
  const int aoff  = hh ? 16 : 0;
  const int boff  = hh ? 16 : 0;

  stage_panel(apanel, xh + (size_t)mt * 32 * D_MODEL, tid);

  const _Float16* W = (which == 0) ? wq : (which == 1) ? wk : wv;
  const _Float16* wb  = W + (size_t)(head * DK + lmod) * D_MODEL + boff;
  const _Float16* lp0 = &apanel[(size_t)lmod * AROW + aoff];
  const _Float16* lp1 = &apanel[(size_t)(lmod + 16) * AROW + aoff];

  f8 acc[2][4];
  gemm_32xk(acc, lp0, lp1, wb);

  const int bidx  = (mt * 32) >> 11;
  const int tbase = (mt * 32) & (T_SEQ - 1);

  if (which < 2) {
    _Float16* dst = (which == 0) ? qh : kh;
    const size_t base_bh = (size_t)(bidx * NHEAD + head) * T_SEQ;
    const bool even = ((lane & 1) == 0);
#pragma unroll
    for (int nt = 0; nt < 4; ++nt) {
      const int nn = nt * 16 + lmod;
      const int i  = nn >> 1;
      const float freq = __expf(-(float)i * 0.28782313662425572f); // 10000^(-i/32)
#pragma unroll
      for (int mi = 0; mi < 2; ++mi)
#pragma unroll
        for (int r = 0; r < 8; ++r) {
          const int trow = tbase + mi * 16 + r + 8 * hh;
          float s, c;
          __sincosf((float)trow * freq, &s, &c);
          float v0 = acc[mi][nt][r];
          float v1 = __shfl_xor(v0, 1, 32);
          float xe = even ? v0 : v1;
          float xo = even ? v1 : v0;
          float res = even ? (xe * c - xo * s) : (xe * s + xo * c);
          dst[(base_bh + trow) * DK + nn] = (_Float16)res;
        }
    }
  } else {
#pragma unroll
    for (int nt = 0; nt < 4; ++nt) {
      const int d = nt * 16 + lmod;
#pragma unroll
      for (int mi = 0; mi < 2; ++mi) {
        h8 pack;
#pragma unroll
        for (int r = 0; r < 8; ++r) pack[r] = (_Float16)acc[mi][nt][r];
        size_t off = ((size_t)(bidx * NHEAD + head) * DK + d) * T_SEQ
                   + tbase + mi * 16 + 8 * hh;
        *(h8*)(vth + off) = pack;
      }
    }
  }
}

// --------------------------- 3) flash attention ----------------------------
__global__ void __launch_bounds__(128)
flash_attn_kernel(const _Float16* __restrict__ qh,
                  const _Float16* __restrict__ kh,
                  const _Float16* __restrict__ vth,
                  _Float16* __restrict__ attnh)
{
  __shared__ __align__(16) _Float16 pbuf[4][16 * 40];

  const int lane = threadIdx.x & 31;
  const int wid  = threadIdx.x >> 5;
  const int g    = blockIdx.x * 4 + wid;
  const int bh   = g >> 7;
  const int qt   = g & 127;
  const int hh   = lane >> 4;
  const int lmod = lane & 15;
  const int aoff = hh ? 16 : 0;
  const int boff = hh ? 16 : 0;

  const _Float16* qp = qh + ((size_t)bh * T_SEQ + qt * 16 + lmod) * DK;
  h16 aq0 = comb16(*(const h8*)(qp + aoff),      *(const h8*)(qp + aoff + 8));
  h16 aq1 = comb16(*(const h8*)(qp + 32 + aoff), *(const h8*)(qp + 32 + aoff + 8));

  float mrow[8], lrow[8];
  f8 o[4];
#pragma unroll
  for (int r = 0; r < 8; ++r) { mrow[r] = -1e30f; lrow[r] = 0.f; }
#pragma unroll
  for (int nt = 0; nt < 4; ++nt) o[nt] = f8zero();

  _Float16* pb = &pbuf[wid][0];
  const _Float16* kbase_p = kh + ((size_t)bh * T_SEQ + lmod) * DK + boff;
  const _Float16* vbase_p = vth + ((size_t)bh * DK + lmod) * T_SEQ + boff;

  const int nchunks = (qt * 16 + 47) >> 5;
  const float scale = 0.125f;   // 1/sqrt(64)

  // ---- prologue: S for chunk 0 ----
  f8 s0, s1;
  {
    const _Float16* kp0 = kbase_p;
    const _Float16* kp1 = kbase_p + (size_t)16 * DK;
    s0 = wmma_f16(aq0, *(const h16*)(kp0),      f8zero());
    s0 = wmma_f16(aq1, *(const h16*)(kp0 + 32), s0);
    s1 = wmma_f16(aq0, *(const h16*)(kp1),      f8zero());
    s1 = wmma_f16(aq1, *(const h16*)(kp1 + 32), s1);
  }

  for (int kc = 0; kc < nchunks; ++kc) {
    const int keybase  = kc * 32;
    // clamped next-chunk base: last iteration re-reads current (discarded)
    const int nextbase = (kc + 1 < nchunks) ? keybase + 32 : keybase;

    // V(kc) loads -- consumed at O stage below
    h16 bv[4];
#pragma unroll
    for (int nt = 0; nt < 4; ++nt)
      bv[nt] = *(const h16*)(vbase_p + (size_t)nt * 16 * T_SEQ + keybase);

    // K(kc+1) loads -- consumed at S stage at loop bottom
    h16 bk[4];
    {
      const _Float16* nk0 = kbase_p + (size_t)nextbase * DK;
      bk[0] = *(const h16*)(nk0);
      bk[1] = *(const h16*)(nk0 + 32);
      bk[2] = *(const h16*)(nk0 + (size_t)16 * DK);
      bk[3] = *(const h16*)(nk0 + (size_t)16 * DK + 32);
    }
    if (kc + 2 < nchunks) {
      __builtin_prefetch(kbase_p + (size_t)(keybase + 64) * DK, 0, 0);
      __builtin_prefetch(kbase_p + (size_t)(keybase + 80) * DK, 0, 0);
    }

    // ---- X(kc): causal mask + scale + chunk row-max ----
    float cm[8];
#pragma unroll
    for (int r = 0; r < 8; ++r) {
      const int tq = qt * 16 + r + 8 * hh;
      const int k0 = keybase + lmod;
      float v0 = (k0      <= tq) ? s0[r] * scale : -1e30f;
      float v1 = (k0 + 16 <= tq) ? s1[r] * scale : -1e30f;
      s0[r] = v0; s1[r] = v1;
      cm[r] = fmaxf(v0, v1);
    }
#pragma unroll
    for (int off = 8; off >= 1; off >>= 1)
#pragma unroll
      for (int r = 0; r < 8; ++r)
        cm[r] = fmaxf(cm[r], __shfl_xor(cm[r], off, 32));

    float alpha[8], rs[8];
#pragma unroll
    for (int r = 0; r < 8; ++r) {
      const float mn = fmaxf(mrow[r], cm[r]);
      alpha[r] = __expf(mrow[r] - mn);
      mrow[r] = mn;
      const float p0 = __expf(s0[r] - mn);
      const float p1 = __expf(s1[r] - mn);
      rs[r] = p0 + p1;
      const int row = r + 8 * hh;
      pb[row * 40 + lmod]      = (_Float16)p0;
      pb[row * 40 + 16 + lmod] = (_Float16)p1;
    }
#pragma unroll
    for (int off = 8; off >= 1; off >>= 1)
#pragma unroll
      for (int r = 0; r < 8; ++r)
        rs[r] += __shfl_xor(rs[r], off, 32);
#pragma unroll
    for (int r = 0; r < 8; ++r) lrow[r] = lrow[r] * alpha[r] + rs[r];
#pragma unroll
    for (int nt = 0; nt < 4; ++nt)
#pragma unroll
      for (int r = 0; r < 8; ++r) o[nt][r] *= alpha[r];

    // wave-private LDS transpose C-layout -> A-layout
    asm volatile("s_wait_dscnt 0" ::: "memory");
    h16 pa = comb16(*(const h8*)(pb + lmod * 40 + aoff),
                    *(const h8*)(pb + lmod * 40 + aoff + 8));

    // ---- O(kc) += P V ----
#pragma unroll
    for (int nt = 0; nt < 4; ++nt)
      o[nt] = wmma_f16(pa, bv[nt], o[nt]);

    // ---- S(kc+1) at loop bottom (garbage on last iter, discarded) ----
    s0 = wmma_f16(aq0, bk[0], f8zero());
    s0 = wmma_f16(aq1, bk[1], s0);
    s1 = wmma_f16(aq0, bk[2], f8zero());
    s1 = wmma_f16(aq1, bk[3], s1);
  }

#pragma unroll
  for (int r = 0; r < 8; ++r) lrow[r] = 1.0f / lrow[r];
  const int b = bh >> 4, h = bh & 15;
#pragma unroll
  for (int nt = 0; nt < 4; ++nt) {
    const int d = nt * 16 + lmod;
#pragma unroll
    for (int r = 0; r < 8; ++r) {
      const int tq = qt * 16 + r + 8 * hh;
      attnh[((size_t)b * T_SEQ + tq) * D_MODEL + h * DK + d] =
          (_Float16)(o[nt][r] * lrow[r]);
    }
  }
}

// --------------------------- 4) output projection --------------------------
__global__ void __launch_bounds__(128)
out_proj_kernel(const _Float16* __restrict__ attnh,
                const _Float16* __restrict__ woh,
                float* __restrict__ out)
{
  __shared__ __align__(16) _Float16 apanel[32 * AROW];

  const int tid  = threadIdx.x;
  const int lane = tid & 31;
  const int wid  = tid >> 5;
  const int mt   = blockIdx.x;
  const int ns   = blockIdx.y * 4 + wid;
  const int hh   = lane >> 4;
  const int lmod = lane & 15;
  const int aoff = hh ? 16 : 0;
  const int boff = hh ? 16 : 0;

  stage_panel(apanel, attnh + (size_t)mt * 32 * D_MODEL, tid);

  const _Float16* wb  = woh + (size_t)(ns * 64 + lmod) * D_MODEL + boff;
  const _Float16* lp0 = &apanel[(size_t)lmod * AROW + aoff];
  const _Float16* lp1 = &apanel[(size_t)(lmod + 16) * AROW + aoff];

  f8 acc[2][4];
  gemm_32xk(acc, lp0, lp1, wb);

#pragma unroll
  for (int nt = 0; nt < 4; ++nt) {
    const int col = ns * 64 + nt * 16 + lmod;
#pragma unroll
    for (int mi = 0; mi < 2; ++mi)
#pragma unroll
      for (int r = 0; r < 8; ++r) {
        const int tok = mt * 32 + mi * 16 + r + 8 * hh;
        out[(size_t)tok * D_MODEL + col] = acc[mi][nt][r];
      }
  }
}

// ------------------------------- launcher ---------------------------------
extern "C" void kernel_launch(void* const* d_in, const int* in_sizes, int n_in,
                              void* d_out, int out_size, void* d_ws, size_t ws_size,
                              hipStream_t stream) {
  const float* x  = (const float*)d_in[0];
  const float* Wq = (const float*)d_in[1];
  const float* Wk = (const float*)d_in[2];
  const float* Wv = (const float*)d_in[3];
  const float* Wo = (const float*)d_in[4];
  float* out = (float*)d_out;

  char* ws = (char*)d_ws;
  size_t off = 0;
  auto alloc = [&](size_t bytes) -> void* {
    void* p = ws + off;
    off += (bytes + 255) & ~(size_t)255;
    return p;
  };

  const size_t nx = (size_t)BT * D_MODEL;
  const size_t nw = (size_t)D_MODEL * D_MODEL;
  _Float16* xh    = (_Float16*)alloc(nx * 2);
  _Float16* wqh   = (_Float16*)alloc(nw * 2);
  _Float16* wkh   = (_Float16*)alloc(nw * 2);
  _Float16* wvh   = (_Float16*)alloc(nw * 2);
  _Float16* woh   = (_Float16*)alloc(nw * 2);
  _Float16* qh    = (_Float16*)alloc(nx * 2);
  _Float16* kh    = (_Float16*)alloc(nx * 2);
  _Float16* vth   = (_Float16*)alloc(nx * 2);
  _Float16* attnh = (_Float16*)alloc(nx * 2);
  (void)ws_size; (void)in_sizes; (void)n_in; (void)out_size;

  cvt_f32_f16<<<(int)(nx / 2048), 256, 0, stream>>>(x,  xh,  (int)nx);
  cvt_f32_f16<<<(int)(nw / 2048), 256, 0, stream>>>(Wq, wqh, (int)nw);
  cvt_f32_f16<<<(int)(nw / 2048), 256, 0, stream>>>(Wk, wkh, (int)nw);
  cvt_f32_f16<<<(int)(nw / 2048), 256, 0, stream>>>(Wv, wvh, (int)nw);
  cvt_f32_f16<<<(int)(nw / 2048), 256, 0, stream>>>(Wo, woh, (int)nw);

  qkv_rope_kernel<<<dim3(BT / 32, 12), 128, 0, stream>>>(
      xh, wqh, wkh, wvh, qh, kh, vth);

  flash_attn_kernel<<<dim3((32 * 128) / 4), 128, 0, stream>>>(
      qh, kh, vth, attnh);

  out_proj_kernel<<<dim3(BT / 32, 4), 128, 0, stream>>>(attnh, woh, out);
}
